// CrossAttentionEAF_7662221656160
// MI455X (gfx1250) — matrix operand order repaired and checked by reading, not compiled, not run on hardware
//
#include <hip/hip_runtime.h>
#include <stdint.h>

typedef __attribute__((ext_vector_type(16))) _Float16 v16h;
typedef __attribute__((ext_vector_type(8)))  _Float16 v8h;
typedef __attribute__((ext_vector_type(8)))  float    v8f;

#define QV    2500
#define QPAD  2512
#define NKTOT 10080
#define SCALE 0.17677669529663687f   // 32^-0.5

// ---------------------------------------------------------------------------
// WMMA helpers (CDNA5 v_wmma_f32_16x16x32_f16).
// A 16x32 f16 layout: lane<16 -> K 0..15, lane>=16 -> K 16..31, M=lane%16
// B 32x16 f16 layout: lane holds column N=lane%16; lanes<16 K=0..15, lanes>=16 K=16..31
// C 16x16 f32 layout: lane holds N=lane%16, rows M = 8*(lane/16)+r for VGPR r
// ---------------------------------------------------------------------------
static __device__ __forceinline__ v16h wload_A(const _Float16* buf, int ld, int rbase, int kbase, int lane){
  const int m  = lane & 15;
  const int ko = (lane < 16) ? 0 : 16;
  const _Float16* rp = buf + (size_t)(rbase + m) * (size_t)ld + kbase;
  v8h lo = *(const v8h*)(rp + ko);        // K = ko..ko+7
  v8h hi = *(const v8h*)(rp + 8 + ko);    // K = ko+8..ko+15
  v16h a;
#pragma unroll
  for (int i = 0; i < 8; ++i){ a[i] = lo[i]; a[i + 8] = hi[i]; }
  return a;
}

// bt is "B transposed": bt[n][k], row stride ld (halves)
static __device__ __forceinline__ v16h wload_B(const _Float16* bt, int ld, int nbase, int kbase, int lane){
  const int n  = lane & 15;
  const int kb = (lane < 16) ? 0 : 16;
  return *(const v16h*)(bt + (size_t)(nbase + n) * (size_t)ld + kbase + kb);
}

static __device__ __forceinline__ v8f wmma16(v16h a, v16h b, v8f c){
  return __builtin_amdgcn_wmma_f32_16x16x32_f16(false, a, false, b, (short)0, c, false, false);
}

// ---------------------------------------------------------------------------
// Weight transpose + f32->f16 convert:  dst[o*K+k] = (f16) src[k*N+o]
// ---------------------------------------------------------------------------
__global__ void k_wcvt(const float* __restrict__ src, _Float16* __restrict__ dst, int K, int N){
  int i = blockIdx.x * 256 + threadIdx.x;
  if (i >= K * N) return;
  int o = i / K;
  int k = i - o * K;
  dst[i] = (_Float16)src[(size_t)k * N + o];
}

// ---------------------------------------------------------------------------
// LayerNorm + QKV projection. One block = 16 rows. 4 waves = 4 x 32 output cols.
// gather_mode: 0 = q stream (src [128][2500]), 1 = k/v stream (src [6][128][1680])
// store_mode : 0 = row-major per head  dst[h][row][32]
//              1 = transposed per head dst[h*32+dh][NKTOT]   (for V)
// ---------------------------------------------------------------------------
__global__ void __launch_bounds__(128)
k_ln_proj(const float* __restrict__ src, const float* __restrict__ lnw, const float* __restrict__ lnb,
          const _Float16* __restrict__ Wt, const float* __restrict__ bias, _Float16* __restrict__ dst,
          int gather_mode, int store_mode, int rows_valid, int ldrows){
  __shared__ float raw[16][128];
  __shared__ __align__(32) _Float16 a_s[16 * 128];
  const int t = threadIdx.x;
  const int j0 = blockIdx.x * 16;

  { // gather + transpose input into LDS (fp32)
    const int r = t >> 3;
    const int dbase = (t & 7) * 16;
    int j = j0 + r; if (j >= rows_valid) j = rows_valid - 1;
    int n = 0, rem = j;
    if (gather_mode == 1){ n = j / 1680; rem = j - n * 1680; }
#pragma unroll 4
    for (int d = dbase; d < dbase + 16; ++d){
      float x;
      if (gather_mode == 0) x = src[(size_t)d * QV + j];
      else                  x = src[(size_t)n * 215040 + (size_t)d * 1680 + rem];
      raw[r][d] = x;
    }
  }
  __syncthreads();
  if (t < 16){ // per-row LayerNorm -> f16 A staging
    float s = 0.f, sq = 0.f;
    for (int d = 0; d < 128; ++d){ float x = raw[t][d]; s += x; sq += x * x; }
    float mean = s * (1.0f / 128.0f);
    float var  = sq * (1.0f / 128.0f) - mean * mean;
    float rs   = rsqrtf(var + 1e-5f);
    for (int d = 0; d < 128; ++d)
      a_s[t * 128 + d] = (_Float16)((raw[t][d] - mean) * rs * lnw[d] + lnb[d]);
  }
  __syncthreads();

  const int lane = t & 31, wv = t >> 5;
  v8f c0 = {}, c1 = {};
#pragma unroll
  for (int kt = 0; kt < 4; ++kt){
    v16h a  = wload_A(a_s, 128, 0, kt * 32, lane);
    v16h b0 = wload_B(Wt, 128, wv * 32,      kt * 32, lane);
    v16h b1 = wload_B(Wt, 128, wv * 32 + 16, kt * 32, lane);
    c0 = wmma16(a, b0, c0);
    c1 = wmma16(a, b1, c1);
  }
  const int n = lane & 15, mb = (lane >> 4) * 8;
#pragma unroll
  for (int tile = 0; tile < 2; ++tile){
    v8f c = tile ? c1 : c0;
    int ncol = wv * 32 + tile * 16 + n;
    int h = ncol >> 5, dh = ncol & 31;
    float bv = bias[ncol];
    if (store_mode == 0){
#pragma unroll
      for (int r = 0; r < 8; ++r)
        dst[((size_t)h * ldrows + (j0 + mb + r)) * 32 + dh] = (_Float16)(c[r] + bv);
    } else {
      v8h pk;
#pragma unroll
      for (int r = 0; r < 8; ++r) pk[r] = (_Float16)(c[r] + bv);
      *(v8h*)(dst + (size_t)(h * 32 + dh) * ldrows + (j0 + mb)) = pk;
    }
  }
}

// ---------------------------------------------------------------------------
// Attention: one block = one 16-row q tile; wave w = head w.
// Streaming over 315 chunks of 32 keys. W_logits/vis are DMA'd into a 3-deep
// LDS ring via global_load_async_to_lds_b128 with two chunks in flight
// (s_wait_asynccnt 2 => oldest chunk has landed, async loads complete in
// order), one workgroup barrier per chunk. K/V tiles prefetched 2 ahead.
// Softmax without running max (logits are O(1)): p = vis ? exp(s*scale*W) : 0.
// ---------------------------------------------------------------------------
__global__ void __launch_bounds__(128)
k_attn(const _Float16* __restrict__ qh, const _Float16* __restrict__ kh,
       const _Float16* __restrict__ vhT, const float* __restrict__ W,
       const int* __restrict__ vis, _Float16* __restrict__ att){
  __shared__ __align__(32) float    w_s[3][512];
  __shared__ __align__(32) int      vis_s[3][512];
  __shared__ __align__(32) _Float16 p_s[4 * 512];

  const int t = threadIdx.x, lane = t & 31, h = t >> 5;
  const int q0 = blockIdx.x * 16;

  const _Float16* qhh = qh  + (size_t)h * QPAD * 32;
  const _Float16* khh = kh  + (size_t)h * NKTOT * 32;
  const _Float16* vhh = vhT + (size_t)h * 32 * NKTOT;

  v16h aq = wload_A(qhh, 32, q0, 0, lane);
  v8f acc0 = {}, acc1 = {};
  float lsum[8] = {0.f,0.f,0.f,0.f,0.f,0.f,0.f,0.f};

  // per-thread staging addresses: 16 rows x 32 cols, 16B per thread
  const int srow = t >> 3, sc4 = (t & 7) * 4;
  int rowg = q0 + srow; if (rowg > QV - 1) rowg = QV - 1;
  const float* wrow = W   + (size_t)rowg * NKTOT;
  const int*   vrow = vis + (size_t)rowg * NKTOT;
  const unsigned ldsW0 = (unsigned)(uintptr_t)&w_s[0][srow * 32 + sc4];
  const unsigned ldsV0 = (unsigned)(uintptr_t)&vis_s[0][srow * 32 + sc4];
  const int n = lane & 15, mb = (lane >> 4) * 8;

#define ISSUE_CHUNK(CH)                                                          \
  {                                                                              \
    int _b = (CH) % 3;                                                           \
    unsigned _lw = ldsW0 + (unsigned)(_b * 2048);                                \
    unsigned _lv = ldsV0 + (unsigned)(_b * 2048);                                \
    unsigned long long _aW = (unsigned long long)(uintptr_t)(wrow + (CH) * 32 + sc4); \
    unsigned long long _aV = (unsigned long long)(uintptr_t)(vrow + (CH) * 32 + sc4); \
    asm volatile("global_load_async_to_lds_b128 %0, %1, off\n\t"                 \
                 "global_load_async_to_lds_b128 %2, %3, off"                     \
                 :: "v"(_lw), "v"(_aW), "v"(_lv), "v"(_aV) : "memory");          \
  }

  ISSUE_CHUNK(0)
  ISSUE_CHUNK(1)

  for (int ch = 0; ch < 315; ++ch){
    const int jb = ch * 32;
    // my two oldest async loads (chunk ch) have landed when <=2 remain in flight
    if (ch < 314) asm volatile("s_wait_asynccnt 2" ::: "memory");
    else          asm volatile("s_wait_asynccnt 0" ::: "memory");
    __syncthreads();               // all waves' chunk-ch pieces landed; prior reads of ring slot done
    if (ch + 2 < 315){
      ISSUE_CHUNK(ch + 2)
      // prefetch K/V tiles two chunks ahead (global_prefetch_b8 path)
      __builtin_prefetch((const void*)(khh + (size_t)(jb + 64 + lane) * 32), 0, 1);
      __builtin_prefetch((const void*)(vhh + (size_t)lane * NKTOT + jb + 64), 0, 1);
    }
    const int buf = ch % 3;
    const float* wsb = w_s[buf];
    const int*   vsb = vis_s[buf];

#pragma unroll
    for (int s = 0; s < 2; ++s){
      v16h bk = wload_B(khh, 32, jb + s * 16, 0, lane);
      v8f  z  = {};
      v8f  S  = wmma16(aq, bk, z);
      const int col = s * 16 + n;
#pragma unroll
      for (int r = 0; r < 8; ++r){
        float wl  = wsb[(mb + r) * 32 + col];
        int   vb  = vsb[(mb + r) * 32 + col];
        float p   = vb ? __expf(S[r] * SCALE * wl) : 0.f;
        lsum[r]  += p;
        p_s[h * 512 + (mb + r) * 32 + col] = (_Float16)p;   // C-layout -> LDS
      }
    }
    asm volatile("s_wait_dscnt 0" ::: "memory");            // P store -> P A-load (same wave)
    v16h ap  = wload_A(p_s + h * 512, 32, 0, 0, lane);      // P as 16x32 A matrix
    v16h bv0 = wload_B(vhh, NKTOT,  0, jb, lane);           // V^T rows = dh
    v16h bv1 = wload_B(vhh, NKTOT, 16, jb, lane);
    acc0 = wmma16(ap, bv0, acc0);
    acc1 = wmma16(ap, bv1, acc1);
  }
#undef ISSUE_CHUNK

  // row-sum reduction across the 16 lanes covering each row half
#pragma unroll
  for (int mask = 8; mask >= 1; mask >>= 1)
#pragma unroll
    for (int r = 0; r < 8; ++r) lsum[r] += __shfl_xor(lsum[r], mask, 32);

#pragma unroll
  for (int r = 0; r < 8; ++r){
    float inv = 1.0f / lsum[r];
    size_t row = (size_t)(q0 + mb + r) * 128 + h * 32;
    att[row + n]      = (_Float16)(acc0[r] * inv);
    att[row + 16 + n] = (_Float16)(acc1[r] * inv);
  }
}

// ---------------------------------------------------------------------------
// Output projection + bias + skip, then pre-LayerNorm -> z (f32 + f16 copies)
// ---------------------------------------------------------------------------
__global__ void __launch_bounds__(128)
k_outproj(const _Float16* __restrict__ att, const _Float16* __restrict__ wpT,
          const float* __restrict__ bp, const float* __restrict__ skip,
          const float* __restrict__ prew, const float* __restrict__ preb,
          float* __restrict__ z32, _Float16* __restrict__ z16){
  __shared__ float zt[16][128];
  const int t = threadIdx.x, lane = t & 31, wv = t >> 5;
  const int q0 = blockIdx.x * 16;
  v8f c0 = {}, c1 = {};
#pragma unroll
  for (int kt = 0; kt < 4; ++kt){
    v16h a  = wload_A(att, 128, q0, kt * 32, lane);
    v16h b0 = wload_B(wpT, 128, wv * 32,      kt * 32, lane);
    v16h b1 = wload_B(wpT, 128, wv * 32 + 16, kt * 32, lane);
    c0 = wmma16(a, b0, c0);
    c1 = wmma16(a, b1, c1);
  }
  const int n = lane & 15, mb = (lane >> 4) * 8;
#pragma unroll
  for (int tile = 0; tile < 2; ++tile){
    v8f c = tile ? c1 : c0;
    int ncol = wv * 32 + tile * 16 + n;
    float bv = bp[ncol];
#pragma unroll
    for (int r = 0; r < 8; ++r){
      int p = q0 + mb + r; int pc = p < QV ? p : QV - 1;
      zt[mb + r][ncol] = c[r] + bv + skip[(size_t)ncol * QV + pc];
    }
  }
  __syncthreads();
  if (t < 16){
    float s = 0.f, sq = 0.f;
    for (int d = 0; d < 128; ++d){ float x = zt[t][d]; s += x; sq += x * x; }
    float mean = s * (1.0f / 128.0f);
    float rs   = rsqrtf(sq * (1.0f / 128.0f) - mean * mean + 1e-5f);
    size_t base = (size_t)(q0 + t) * 128;
    for (int d = 0; d < 128; ++d){
      float y = (zt[t][d] - mean) * rs * prew[d] + preb[d];
      z32[base + d] = y;
      z16[base + d] = (_Float16)y;
    }
  }
}

// ---------------------------------------------------------------------------
// FFN layer 1: h1 = gelu_exact(z @ w1 + b1), K=128, N=256 (each wave 64 cols)
// ---------------------------------------------------------------------------
__global__ void __launch_bounds__(128)
k_ffn1(const _Float16* __restrict__ z16, const _Float16* __restrict__ w1T,
       const float* __restrict__ b1, _Float16* __restrict__ h1){
  const int t = threadIdx.x, lane = t & 31, wv = t >> 5;
  const int q0 = blockIdx.x * 16;
  v8f c[4] = {};
#pragma unroll
  for (int kt = 0; kt < 4; ++kt){
    v16h a = wload_A(z16, 128, q0, kt * 32, lane);
#pragma unroll
    for (int ti = 0; ti < 4; ++ti){
      v16h b = wload_B(w1T, 128, wv * 64 + ti * 16, kt * 32, lane);
      c[ti] = wmma16(a, b, c[ti]);
    }
  }
  const int n = lane & 15, mb = (lane >> 4) * 8;
#pragma unroll
  for (int ti = 0; ti < 4; ++ti){
    int ncol = wv * 64 + ti * 16 + n;
    float bv = b1[ncol];
#pragma unroll
    for (int r = 0; r < 8; ++r){
      float x = c[ti][r] + bv;
      float g = 0.5f * x * (1.0f + erff(x * 0.70710678118654752f));  // exact gelu
      h1[(size_t)(q0 + mb + r) * 256 + ncol] = (_Float16)g;
    }
  }
}

// ---------------------------------------------------------------------------
// FFN layer 2 + residual + post-LayerNorm + final transpose store [D][2500]
// ---------------------------------------------------------------------------
__global__ void __launch_bounds__(128)
k_ffn2(const _Float16* __restrict__ h1, const _Float16* __restrict__ w2T,
       const float* __restrict__ b2, const float* __restrict__ z32,
       const float* __restrict__ postw, const float* __restrict__ postb,
       float* __restrict__ out){
  __shared__ float zt[16][128];
  const int t = threadIdx.x, lane = t & 31, wv = t >> 5;
  const int q0 = blockIdx.x * 16;
  v8f c0 = {}, c1 = {};
#pragma unroll
  for (int kt = 0; kt < 8; ++kt){
    v16h a  = wload_A(h1, 256, q0, kt * 32, lane);
    v16h b0 = wload_B(w2T, 256, wv * 32,      kt * 32, lane);
    v16h b1 = wload_B(w2T, 256, wv * 32 + 16, kt * 32, lane);
    c0 = wmma16(a, b0, c0);
    c1 = wmma16(a, b1, c1);
  }
  const int n = lane & 15, mb = (lane >> 4) * 8;
#pragma unroll
  for (int tile = 0; tile < 2; ++tile){
    v8f c = tile ? c1 : c0;
    int ncol = wv * 32 + tile * 16 + n;
    float bv = b2[ncol];
#pragma unroll
    for (int r = 0; r < 8; ++r){
      int p = q0 + mb + r;
      zt[mb + r][ncol] = c[r] + bv + z32[(size_t)p * 128 + ncol];
    }
  }
  __syncthreads();
  if (t < 16){
    int p = q0 + t;
    if (p < QV){
      float s = 0.f, sq = 0.f;
      for (int d = 0; d < 128; ++d){ float x = zt[t][d]; s += x; sq += x * x; }
      float mean = s * (1.0f / 128.0f);
      float rs   = rsqrtf(sq * (1.0f / 128.0f) - mean * mean + 1e-5f);
      for (int d = 0; d < 128; ++d)
        out[(size_t)d * QV + p] = (zt[t][d] - mean) * rs * postw[d] + postb[d];
    }
  }
}

// ---------------------------------------------------------------------------
extern "C" void kernel_launch(void* const* d_in, const int* in_sizes, int n_in,
                              void* d_out, int out_size, void* d_ws, size_t ws_size,
                              hipStream_t stream){
  const float* q    = (const float*)d_in[0];
  const float* k    = (const float*)d_in[1];
  const float* v    = (const float*)d_in[2];
  const float* Wl   = (const float*)d_in[3];
  const int*   vis  = (const int*)  d_in[4];
  const float* skip = (const float*)d_in[5];
  const float* qn_w = (const float*)d_in[6];  const float* qn_b = (const float*)d_in[7];
  const float* kn_w = (const float*)d_in[8];  const float* kn_b = (const float*)d_in[9];
  const float* vn_w = (const float*)d_in[10]; const float* vn_b = (const float*)d_in[11];
  const float* pre_w = (const float*)d_in[12]; const float* pre_b = (const float*)d_in[13];
  const float* post_w = (const float*)d_in[14]; const float* post_b = (const float*)d_in[15];
  const float* wq = (const float*)d_in[16]; const float* bq = (const float*)d_in[17];
  const float* wk = (const float*)d_in[18]; const float* bk = (const float*)d_in[19];
  const float* wv = (const float*)d_in[20]; const float* bv = (const float*)d_in[21];
  const float* wp = (const float*)d_in[22]; const float* bp = (const float*)d_in[23];
  const float* w1 = (const float*)d_in[24]; const float* b1 = (const float*)d_in[25];
  const float* w2 = (const float*)d_in[26]; const float* b2 = (const float*)d_in[27];
  (void)in_sizes; (void)n_in; (void)out_size; (void)ws_size;

  char* ws = (char*)d_ws;
  _Float16* wqT = (_Float16*)(ws + 0);         // 128x128
  _Float16* wkT = (_Float16*)(ws + 32768);
  _Float16* wvT = (_Float16*)(ws + 65536);
  _Float16* wpT = (_Float16*)(ws + 98304);
  _Float16* w1T = (_Float16*)(ws + 131072);    // 256x128
  _Float16* w2T = (_Float16*)(ws + 196608);    // 128x256
  _Float16* qh  = (_Float16*)(ws + 262144);    // 4 x 2512 x 32
  _Float16* kh  = (_Float16*)(ws + 905216);    // 4 x 10080 x 32
  _Float16* vhT = (_Float16*)(ws + 3485696);   // 4 x 32 x 10080
  _Float16* att = (_Float16*)(ws + 6066176);   // 2512 x 128
  float*    z32 = (float*)   (ws + 6709248);   // 2512 x 128
  _Float16* z16 = (_Float16*)(ws + 7995392);   // 2512 x 128
  _Float16* h1  = (_Float16*)(ws + 8638464);   // 2512 x 256

  // weight transpose + fp16 conversion
  k_wcvt<<<64,  256, 0, stream>>>(wq, wqT, 128, 128);
  k_wcvt<<<64,  256, 0, stream>>>(wk, wkT, 128, 128);
  k_wcvt<<<64,  256, 0, stream>>>(wv, wvT, 128, 128);
  k_wcvt<<<64,  256, 0, stream>>>(wp, wpT, 128, 128);
  k_wcvt<<<128, 256, 0, stream>>>(w1, w1T, 128, 256);
  k_wcvt<<<128, 256, 0, stream>>>(w2, w2T, 256, 128);

  // LayerNorm + projections
  k_ln_proj<<<QPAD / 16,  128, 0, stream>>>(q, qn_w, qn_b, wqT, bq, qh,  0, 0, QV,    QPAD);
  k_ln_proj<<<NKTOT / 16, 128, 0, stream>>>(k, kn_w, kn_b, wkT, bk, kh,  1, 0, NKTOT, NKTOT);
  k_ln_proj<<<NKTOT / 16, 128, 0, stream>>>(v, vn_w, vn_b, wvT, bv, vhT, 1, 1, NKTOT, NKTOT);

  // attention (4 heads per block, pipelined async-LDS staged W/vis)
  k_attn<<<QPAD / 16, 128, 0, stream>>>(qh, kh, vhT, Wl, vis, att);

  // output projection + skip + pre-LN
  k_outproj<<<QPAD / 16, 128, 0, stream>>>(att, wpT, bp, skip, pre_w, pre_b, z32, z16);

  // FFN + post-LN + transpose out
  k_ffn1<<<QPAD / 16, 128, 0, stream>>>(z16, w1T, b1, h1);
  k_ffn2<<<QPAD / 16, 128, 0, stream>>>(h1, w2T, b2, z32, post_w, post_b, (float*)d_out);
}
